// GCN_18820546691088
// MI455X (gfx1250) — matrix operand order repaired and checked by reading, not compiled, not run on hardware
//
#include <hip/hip_runtime.h>

// ---------------- problem constants (from reference) ----------------
#define NUM_NODES 50000
#define NUM_EDGES 800000
#define NUM_LABEL_EDGES 100000
#define F_IN 128
#define F_HID 128
#define F_OUT 64

// ---------------- WMMA types (probe-confirmed signatures) ----------------
typedef __attribute__((ext_vector_type(16))) __bf16 v16bf;
typedef __attribute__((ext_vector_type(8)))  float  v8f;

__device__ __forceinline__ unsigned short f2bf(float f) {
    // round-to-nearest-even fp32 -> bf16 (bit-level, no __bf16 conversion needed)
    unsigned u = __float_as_uint(f);
    unsigned r = (u + 0x7FFFu + ((u >> 16) & 1u)) >> 16;
    return (unsigned short)r;
}

// LDS row stride in bf16 elements, padded 128->130 to avoid bank conflicts
#define LDSS 130

// ---------------------------------------------------------------------
// GEMM: H[M,N] = cvt_bf16(maybe_relu(X[M,128])) @ cvt_bf16(W[128,N])
// block = 128 threads (4 waves); block tile = 16 (M) x 64 (N); K = 128.
// grid = (M/16, N/64)
// ---------------------------------------------------------------------
__global__ __launch_bounds__(128)
void gemm_bf16_wmma(const float* __restrict__ X, const float* __restrict__ W,
                    float* __restrict__ H, int N, int applyRelu)
{
    __shared__ unsigned short sA[16 * LDSS];   // row-major, rows 0..15, k 0..127
    __shared__ unsigned short sB[64 * LDSS];   // column-major, cols 0..63, k 0..127

    const int tid   = threadIdx.x;
    const int lane  = tid & 31;
    const int wave  = tid >> 5;
    const int tileM = blockIdx.x;           // 16 rows per tile
    const int n0    = blockIdx.y * 64;      // 64 cols per block

    // ---- stage A panel (16 x 128), fp32 -> bf16, relu folded in ----
    {
        const float* src = X + (size_t)tileM * 16 * 128;
        #pragma unroll
        for (int i = tid; i < 16 * 128; i += 128) {
            int m = i >> 7, k = i & 127;
            float v = src[i];                       // coalesced
            if (applyRelu) v = fmaxf(v, 0.0f);
            sA[m * LDSS + k] = f2bf(v);
        }
    }
    // ---- stage B panel (128 x 64) column-major into LDS ----
    {
        #pragma unroll
        for (int i = tid; i < 128 * 64; i += 128) {
            int k = i >> 6, nl = i & 63;
            float v = W[(size_t)k * N + n0 + nl];   // coalesced over nl
            sB[nl * LDSS + k] = f2bf(v);
        }
    }
    __syncthreads();

    // ---- each wave: one 16x16 tile at columns n0 + wave*16 ----
    const int nw     = wave * 16;
    const int mrow   = lane & 15;        // row (A) / col (B) index for this lane
    const int hiHalf = lane >> 4;        // lane group 0..15 vs 16..31

    v8f acc = {0.f, 0.f, 0.f, 0.f, 0.f, 0.f, 0.f, 0.f};

    #pragma unroll
    for (int kk = 0; kk < 128; kk += 32) {
        union { unsigned u[8]; v16bf v; } fa, fb;

        // A fragment (16x32 bf16): lanes 0-15 hold K = kk+[0..7], kk+[16..23];
        // lanes 16-31 hold K = kk+[8..15], kk+[24..31]; 2 bf16 per dword.
        const unsigned* arow = (const unsigned*)&sA[mrow * LDSS];
        const int akLo = (kk + hiHalf * 8) >> 1;        // dword index of K pair
        #pragma unroll
        for (int v = 0; v < 4; ++v) fa.u[v]     = arow[akLo + v];
        #pragma unroll
        for (int v = 0; v < 4; ++v) fa.u[4 + v] = arow[akLo + 8 + v];

        // B fragment (32x16 bf16): lane holds column (nw+mrow);
        // lanes 0-15: K = kk+[0..7], kk+[16..23]; lanes 16-31: K = kk+[8..15], kk+[24..31].
        const unsigned* bcol = (const unsigned*)&sB[(nw + mrow) * LDSS];
        const int bkLo = (kk + hiHalf * 8) >> 1;
        #pragma unroll
        for (int v = 0; v < 4; ++v) fb.u[v] = bcol[bkLo + v];
        #pragma unroll
        for (int v = 0; v < 4; ++v) fb.u[4 + v] = bcol[bkLo + 8 + v];

        acc = __builtin_amdgcn_wmma_f32_16x16x32_bf16(
            /*neg_a=*/false, fa.v, /*neg_b=*/false, fb.v,
            /*c_mod=*/(short)0, acc, /*reuse_a=*/false, /*reuse_b=*/false);
    }

    // ---- store C: VGPR r, lanes 0-15 -> row r, lanes 16-31 -> row r+8 ----
    float* out = H + (size_t)tileM * 16 * N + n0 + nw;
    #pragma unroll
    for (int r = 0; r < 8; ++r) {
        int m = r + hiHalf * 8;
        out[(size_t)m * N + mrow] = acc[r];
    }
}

// ---------------- degree / normalization ----------------
__global__ void deg_init(float* deg, int n) {
    int i = blockIdx.x * blockDim.x + threadIdx.x;
    if (i < n) deg[i] = 1.0f;                    // self-loop
}
__global__ void deg_count(const int* __restrict__ dst, float* deg, int E) {
    int e = blockIdx.x * blockDim.x + threadIdx.x;
    if (e < E) atomicAdd(&deg[dst[e]], 1.0f);
}
__global__ void deg_rsqrt(float* deg, int n) {
    int i = blockIdx.x * blockDim.x + threadIdx.x;
    if (i < n) deg[i] = rsqrtf(deg[i]);          // dinv
}

// ---------------- A = H * dinv^2 + bias (self-loop + bias init) ----------------
__global__ void selfloop_bias(const float* __restrict__ H, const float* __restrict__ dinv,
                              const float* __restrict__ bias, float* __restrict__ A,
                              int n, int F)
{
    int i = blockIdx.x * blockDim.x + threadIdx.x;
    if (i < n * F) {
        int node = i / F, f = i - node * F;
        float di = dinv[node];
        A[i] = H[i] * di * di + bias[f];
    }
}

// ---------------- edge-parallel aggregation: one wave per edge ----------------
__global__ void aggregate(const float* __restrict__ H, const float* __restrict__ dinv,
                          const int* __restrict__ src, const int* __restrict__ dst,
                          float* __restrict__ A, int E, int F)
{
    int e    = (blockIdx.x * blockDim.x + threadIdx.x) >> 5;
    int lane = threadIdx.x & 31;
    if (e >= E) return;
    int s = src[e], d = dst[e];
    float norm = dinv[s] * dinv[d];
    const float* hs = H + (size_t)s * F;
    float*       ad = A + (size_t)d * F;
    for (int f = lane; f < F; f += 32)
        atomicAdd(&ad[f], hs[f] * norm);         // L2-resident atomics
}

// ---------------- decode: per-label-edge dot product, wave reduction ----------------
__global__ void decode(const float* __restrict__ Z, const int* __restrict__ la,
                       const int* __restrict__ lb, float* __restrict__ out, int E, int F)
{
    int e    = (blockIdx.x * blockDim.x + threadIdx.x) >> 5;
    int lane = threadIdx.x & 31;
    if (e >= E) return;
    const float* za = Z + (size_t)la[e] * F;
    const float* zb = Z + (size_t)lb[e] * F;
    float s = 0.f;
    for (int f = lane; f < F; f += 32) s += za[f] * zb[f];
    #pragma unroll
    for (int off = 16; off; off >>= 1) s += __shfl_down(s, off, 32);
    if (lane == 0) out[e] = s;
}

// ---------------------------------------------------------------------
extern "C" void kernel_launch(void* const* d_in, const int* in_sizes, int n_in,
                              void* d_out, int out_size, void* d_ws, size_t ws_size,
                              hipStream_t stream)
{
    const float* x   = (const float*)d_in[0];
    const int*   ei  = (const int*)d_in[1];           // [2, E]
    const int*   eli = (const int*)d_in[2];           // [2, LE]
    const float* W1  = (const float*)d_in[3];
    const float* b1  = (const float*)d_in[4];
    const float* W2  = (const float*)d_in[5];
    const float* b2  = (const float*)d_in[6];
    const float* W3  = (const float*)d_in[7];
    const float* b3  = (const float*)d_in[8];
    float* out = (float*)d_out;

    const int* src = ei;
    const int* dst = ei + NUM_EDGES;
    const int* la  = eli;
    const int* lb  = eli + NUM_LABEL_EDGES;

    // workspace layout (floats): dinv | Hbuf | Abuf
    float* ws   = (float*)d_ws;
    float* dinv = ws;
    float* Hbuf = ws + 50048;                          // 64B-aligned past dinv
    float* Abuf = Hbuf + (size_t)NUM_NODES * 128;

    const int T = 256;
    dim3 blkNode((NUM_NODES + T - 1) / T);
    dim3 blkEdge((NUM_EDGES + T - 1) / T);
    dim3 blkEdgeWave((NUM_EDGES * 32 + T - 1) / T);    // wave per edge
    dim3 blkLblWave((NUM_LABEL_EDGES * 32 + T - 1) / T);

    // 1) symmetric-normalization coefficients
    deg_init <<<blkNode, T, 0, stream>>>(dinv, NUM_NODES);
    deg_count<<<blkEdge, T, 0, stream>>>(dst, dinv, NUM_EDGES);
    deg_rsqrt<<<blkNode, T, 0, stream>>>(dinv, NUM_NODES);

    const int MT = NUM_NODES / 16;                     // 3125 exact

    // 2) layer 1: H = x @ W1 ; A = H*dinv^2 + b1 ; A += scatter
    gemm_bf16_wmma<<<dim3(MT, F_HID / 64), 128, 0, stream>>>(x, W1, Hbuf, F_HID, 0);
    selfloop_bias<<<dim3((NUM_NODES * F_HID + T - 1) / T), T, 0, stream>>>(Hbuf, dinv, b1, Abuf, NUM_NODES, F_HID);
    aggregate<<<blkEdgeWave, T, 0, stream>>>(Hbuf, dinv, src, dst, Abuf, NUM_EDGES, F_HID);

    // 3) layer 2 (relu of layer-1 output folded into GEMM A-load)
    gemm_bf16_wmma<<<dim3(MT, F_HID / 64), 128, 0, stream>>>(Abuf, W2, Hbuf, F_HID, 1);
    selfloop_bias<<<dim3((NUM_NODES * F_HID + T - 1) / T), T, 0, stream>>>(Hbuf, dinv, b2, Abuf, NUM_NODES, F_HID);
    aggregate<<<blkEdgeWave, T, 0, stream>>>(Hbuf, dinv, src, dst, Abuf, NUM_EDGES, F_HID);

    // 4) layer 3: 128 -> 64, relu folded, no relu on output
    gemm_bf16_wmma<<<dim3(MT, F_OUT / 64), 128, 0, stream>>>(Abuf, W3, Hbuf, F_OUT, 1);
    selfloop_bias<<<dim3((NUM_NODES * F_OUT + T - 1) / T), T, 0, stream>>>(Hbuf, dinv, b3, Abuf, NUM_NODES, F_OUT);
    aggregate<<<blkEdgeWave, T, 0, stream>>>(Hbuf, dinv, src, dst, Abuf, NUM_EDGES, F_OUT);

    // 5) decode: per-label-edge dot product of 64-dim embeddings
    decode<<<blkLblWave, T, 0, stream>>>(Abuf, la, lb, out, NUM_LABEL_EDGES, F_OUT);
}
